// GlobalModel_10763188043966
// MI455X (gfx1250) — matrix operand order repaired and checked by reading, not compiled, not run on hardware
//
#include <hip/hip_runtime.h>
#include <hip/hip_bf16.h>

typedef __attribute__((ext_vector_type(16))) __bf16 v16bf;
typedef __attribute__((ext_vector_type(8)))  float  v8f;

#define N_NODES 40000
#define N_EDGES 640000
#define DIM_H   512
#define N_HEADS 8
#define D_HEAD  64
#define NQKV    1536   // Q | K | V concatenated per row

// ---------------- workspace layout (bytes, all 256-aligned) ----------------
static constexpr size_t OFF_WT     = 0;                                        // bf16 [1536][512]
static constexpr size_t OFF_QKV    = 1572864;                                  // bf16 [40000][1536]
static constexpr size_t OFF_EBIN   = OFF_QKV  + (size_t)N_NODES * NQKV * 2;    // int  [640000] src, dst-binned
static constexpr size_t OFF_DEG    = OFF_EBIN + (size_t)N_EDGES * 4;           // int  [40000]
static constexpr size_t OFF_OFFS   = OFF_DEG  + 160000;                        // int  [40001] (padded)
static constexpr size_t OFF_CURSOR = OFF_OFFS + 160256;                        // int  [40000]
static constexpr size_t OFF_COLSUM = OFF_CURSOR + 160000;                      // f32 [512]
static constexpr size_t OFF_COLSQ  = OFF_COLSUM + 2048;
static constexpr size_t OFF_SCALE  = OFF_COLSQ  + 2048;
static constexpr size_t OFF_SHIFT  = OFF_SCALE  + 2048;

// ---------------- pack weights: wt[c][k] = W(sel)[k][c&511] as bf16 ----------------
__global__ __launch_bounds__(256)
void pack_wt(const float* __restrict__ Wq, const float* __restrict__ Wk,
             const float* __restrict__ Wv, __bf16* __restrict__ wt) {
    int idx = blockIdx.x * 256 + threadIdx.x;
    if (idx >= NQKV * DIM_H) return;
    int c = idx / DIM_H;
    int k = idx - c * DIM_H;
    const float* W = (c < 512) ? Wq : (c < 1024 ? Wk : Wv);
    wt[idx] = (__bf16)W[(size_t)k * DIM_H + (c & 511)];
}

// Build a contiguous-k (k = 16*hi + s) 16x32 bf16 A-fragment from an fp32 row (lane half = hi).
__device__ __forceinline__ v16bf load_a_frag(const float* __restrict__ xr, int k0, int hi) {
    const int b1 = k0 + hi * 16;
    const int b2 = k0 + hi * 16 + 8;
    float4 fa0 = *reinterpret_cast<const float4*>(xr + b1);
    float4 fa1 = *reinterpret_cast<const float4*>(xr + b1 + 4);
    float4 fb0 = *reinterpret_cast<const float4*>(xr + b2);
    float4 fb1 = *reinterpret_cast<const float4*>(xr + b2 + 4);
    v16bf a;
    a[0]=(__bf16)fa0.x; a[1]=(__bf16)fa0.y; a[2]=(__bf16)fa0.z; a[3]=(__bf16)fa0.w;
    a[4]=(__bf16)fa1.x; a[5]=(__bf16)fa1.y; a[6]=(__bf16)fa1.z; a[7]=(__bf16)fa1.w;
    a[8]=(__bf16)fb0.x; a[9]=(__bf16)fb0.y; a[10]=(__bf16)fb0.z; a[11]=(__bf16)fb0.w;
    a[12]=(__bf16)fb1.x; a[13]=(__bf16)fb1.y; a[14]=(__bf16)fb1.z; a[15]=(__bf16)fb1.w;
    return a;
}

// ---------------- fused QKV GEMM: one wave32 computes a 32x64 tile ----------------
// 8 v_wmma_f32_16x16x32_bf16 per k-step; B fragments reused for both row halves.
__global__ __launch_bounds__(256)
void qkv_gemm(const float* __restrict__ x, const __bf16* __restrict__ wt,
              __bf16* __restrict__ qkv) {
    const int tid   = threadIdx.x;
    const int lane  = tid & 31;
    const int wave  = blockIdx.x * 8 + (tid >> 5);
    const int NT_N  = NQKV / 64;              // 24 column tiles
    const int mtile = wave / NT_N;            // 0..1249 (32-row tiles)
    const int ntile = wave - mtile * NT_N;
    if (mtile >= N_NODES / 32) return;        // uniform per wave

    const int r    = lane & 15;
    const int hi   = lane >> 4;
    const int row0 = mtile * 32;
    const int col0 = ntile * 64;

    const v8f vzero = {0.f,0.f,0.f,0.f,0.f,0.f,0.f,0.f};
    v8f acc[2][4];
    #pragma unroll
    for (int h = 0; h < 2; ++h)
        #pragma unroll
        for (int t = 0; t < 4; ++t) acc[h][t] = vzero;

    const float* xr0 = x + (size_t)(row0 + r) * DIM_H;
    const float* xr1 = x + (size_t)(row0 + 16 + r) * DIM_H;

    for (int k0 = 0; k0 < DIM_H; k0 += 32) {
        v16bf a0 = load_a_frag(xr0, k0, hi);
        v16bf a1 = load_a_frag(xr1, k0, hi);
        const int kb = k0 + hi * 16;          // B: 16 contiguous K rows per lane half
        #pragma unroll
        for (int t = 0; t < 4; ++t) {
            const __bf16* bp = wt + (size_t)(col0 + t * 16 + r) * DIM_H + kb;
            v16bf b = *reinterpret_cast<const v16bf*>(bp);
            acc[0][t] = __builtin_amdgcn_wmma_f32_16x16x32_bf16(
                false, a0, false, b, (short)0, acc[0][t], false, false);
            acc[1][t] = __builtin_amdgcn_wmma_f32_16x16x32_bf16(
                false, a1, false, b, (short)0, acc[1][t], false, false);
        }
    }

    #pragma unroll
    for (int h = 0; h < 2; ++h)
        #pragma unroll
        for (int t = 0; t < 4; ++t) {
            const int col = col0 + t * 16 + r;
            #pragma unroll
            for (int i = 0; i < 8; ++i) {
                const int row = row0 + h * 16 + hi * 8 + i;
                qkv[(size_t)row * NQKV + col] = (__bf16)acc[h][t][i];
            }
        }
}

// ---------------- counting sort of edges by destination ----------------
__global__ __launch_bounds__(256)
void count_deg(const int* __restrict__ ei, int* __restrict__ deg) {
    int e = blockIdx.x * 256 + threadIdx.x;
    if (e < N_EDGES) atomicAdd(&deg[ei[N_EDGES + e]], 1);
}

__global__ __launch_bounds__(1024)
void scan_offsets(const int* __restrict__ deg, int* __restrict__ offs,
                  int* __restrict__ cursor) {
    __shared__ int partial[1024];
    const int t = threadIdx.x;
    const int CHUNK = 40;                     // 1024*40 >= 40000
    int local = 0;
    for (int j = 0; j < CHUNK; ++j) {
        int idx = t * CHUNK + j;
        if (idx < N_NODES) local += deg[idx];
    }
    partial[t] = local;
    __syncthreads();
    for (int off = 1; off < 1024; off <<= 1) {   // Hillis-Steele inclusive scan
        int v = (t >= off) ? partial[t - off] : 0;
        __syncthreads();
        partial[t] += v;
        __syncthreads();
    }
    int prefix = (t == 0) ? 0 : partial[t - 1];  // exclusive prefix
    for (int j = 0; j < CHUNK; ++j) {
        int idx = t * CHUNK + j;
        if (idx < N_NODES) {
            offs[idx]   = prefix;
            cursor[idx] = prefix;
            prefix += deg[idx];
        }
    }
    if (t == 1023) offs[N_NODES] = partial[1023];
}

__global__ __launch_bounds__(256)
void bin_edges(const int* __restrict__ ei, int* __restrict__ cursor,
               int* __restrict__ ebin) {
    int e = blockIdx.x * 256 + threadIdx.x;
    if (e < N_EDGES) {
        int dst = ei[N_EDGES + e];
        int pos = atomicAdd(&cursor[dst], 1);
        ebin[pos] = ei[e];                    // store src node id
    }
}

// ---------------- gather attention: one wave32 per destination node ----------------
// lane l owns dims [16l,16l+16); head = l>>2. Q and accumulators live in registers;
// K/V gathers hit the L2-resident bf16 QKV buffer. Zero scatter atomics.
__global__ __launch_bounds__(256)
void node_attn(const __bf16* __restrict__ qkv, const float* __restrict__ x,
               const int* __restrict__ offs, const int* __restrict__ ebin,
               float* __restrict__ h_out) {
    const int n    = (blockIdx.x * 256 + threadIdx.x) >> 5;
    const int lane = threadIdx.x & 31;
    if (n >= N_NODES) return;                 // uniform per wave

    v16bf qv = *reinterpret_cast<const v16bf*>(qkv + (size_t)n * NQKV + lane * 16);
    float q[16];
    #pragma unroll
    for (int j = 0; j < 16; ++j) q[j] = (float)qv[j];

    float wv[16];
    #pragma unroll
    for (int j = 0; j < 16; ++j) wv[j] = 0.f;
    float z = 0.f;

    const int e0 = offs[n], e1 = offs[n + 1];
    for (int e = e0; e < e1; ++e) {
        const int src = ebin[e];
        if (e + 1 < e1) {                     // prefetch next K row (global_prefetch_b8)
            __builtin_prefetch(qkv + (size_t)ebin[e + 1] * NQKV + 512 + lane * 16, 0, 0);
        }
        const __bf16* kb = qkv + (size_t)src * NQKV +  512 + lane * 16;
        const __bf16* vb = qkv + (size_t)src * NQKV + 1024 + lane * 16;
        v16bf kv = *reinterpret_cast<const v16bf*>(kb);
        v16bf vv = *reinterpret_cast<const v16bf*>(vb);

        float p = 0.f;
        #pragma unroll
        for (int j = 0; j < 16; ++j) p += q[j] * (float)kv[j];
        p += __shfl_xor(p, 1, 32);            // reduce 4 lanes of this head
        p += __shfl_xor(p, 2, 32);

        float s = __expf(fminf(fmaxf(p * 0.125f, -5.f), 5.f));
        #pragma unroll
        for (int j = 0; j < 16; ++j) wv[j] += (float)vv[j] * s;
        z += s;                               // all 4 lanes of a head hold identical s
    }

    const float invz = 1.0f / (z + 1e-6f);
    const float* xr = x + (size_t)n * DIM_H + lane * 16;
    float*       hr = h_out + (size_t)n * DIM_H + lane * 16;
    #pragma unroll
    for (int j = 0; j < 16; ++j) hr[j] = xr[j] + wv[j] * invz;   // coalesced b128 stores
}

// ---------------- per-column partial stats over h ----------------
__global__ __launch_bounds__(512)
void col_stats(const float* __restrict__ h, float* __restrict__ colsum,
               float* __restrict__ colsq) {
    const int c  = threadIdx.x;
    const int n0 = blockIdx.x * 50;           // 800 blocks x 50 rows
    float s = 0.f, ss = 0.f;
    for (int n = n0; n < n0 + 50; ++n) {
        float hv = h[(size_t)n * DIM_H + c];
        s += hv; ss += hv * hv;
    }
    atomicAdd(&colsum[c], s);
    atomicAdd(&colsq[c], ss);
}

__global__ __launch_bounds__(512)
void bn_stats(const float* __restrict__ colsum, const float* __restrict__ colsq,
              const float* __restrict__ gamma, const float* __restrict__ beta,
              float* __restrict__ scale, float* __restrict__ shift) {
    const int c = threadIdx.x;
    const float inv_n = 1.0f / (float)N_NODES;
    float mean = colsum[c] * inv_n;
    float var  = colsq[c] * inv_n - mean * mean;    // biased variance
    float rstd = rsqrtf(var + 1e-5f);
    float sc   = rstd * gamma[c];
    scale[c] = sc;
    shift[c] = beta[c] - mean * sc;
}

__global__ __launch_bounds__(256)
void bn_apply(float* __restrict__ out, const float* __restrict__ scale,
              const float* __restrict__ shift) {
    int idx = blockIdx.x * 256 + threadIdx.x;
    if (idx >= N_NODES * DIM_H) return;
    int c = idx & 511;
    out[idx] = out[idx] * scale[c] + shift[c];
}

extern "C" void kernel_launch(void* const* d_in, const int* in_sizes, int n_in,
                              void* d_out, int out_size, void* d_ws, size_t ws_size,
                              hipStream_t stream) {
    const float* x     = (const float*)d_in[0];
    const int*   ei    = (const int*)d_in[1];
    // d_in[2] virt_h, d_in[3] virt_edge_index: unused (use_virt_nodes=False)
    const float* Wq    = (const float*)d_in[4];
    const float* Wk    = (const float*)d_in[5];
    const float* Wv    = (const float*)d_in[6];
    const float* gamma = (const float*)d_in[7];
    const float* beta  = (const float*)d_in[8];
    float* out = (float*)d_out;

    char* ws = (char*)d_ws;
    __bf16* wt     = (__bf16*)(ws + OFF_WT);
    __bf16* qkv    = (__bf16*)(ws + OFF_QKV);
    int*    ebin   = (int*)(ws + OFF_EBIN);
    int*    deg    = (int*)(ws + OFF_DEG);
    int*    offs   = (int*)(ws + OFF_OFFS);
    int*    cursor = (int*)(ws + OFF_CURSOR);
    float*  colsum = (float*)(ws + OFF_COLSUM);
    float*  colsq  = (float*)(ws + OFF_COLSQ);
    float*  scale  = (float*)(ws + OFF_SCALE);
    float*  shift  = (float*)(ws + OFF_SHIFT);

    // zero accumulation buffers (graph-capture-safe stream ops)
    hipMemsetAsync(deg, 0, (size_t)N_NODES * 4, stream);
    hipMemsetAsync(colsum, 0, 2 * 512 * 4, stream);   // colsum + colsq contiguous

    // 1) transpose+convert weights to bf16
    pack_wt<<<(NQKV * DIM_H + 255) / 256, 256, 0, stream>>>(Wq, Wk, Wv, wt);

    // 2) fused QKV GEMM via bf16 WMMA: 1250 x 24 tiles = 30000 waves
    qkv_gemm<<<(1250 * 24) / 8, 256, 0, stream>>>(x, wt, qkv);

    // 3) counting-sort edges by destination
    count_deg  <<<(N_EDGES + 255) / 256, 256, 0, stream>>>(ei, deg);
    scan_offsets<<<1, 1024, 0, stream>>>(deg, offs, cursor);
    bin_edges  <<<(N_EDGES + 255) / 256, 256, 0, stream>>>(ei, cursor, ebin);

    // 4) gather attention + residual, one wave per node (no scatter atomics)
    node_attn<<<(N_NODES * 32) / 256, 256, 0, stream>>>(qkv, x, offs, ebin, out);

    // 5) BN: column stats -> affine -> in-place apply
    col_stats<<<N_NODES / 50, 512, 0, stream>>>(out, colsum, colsq);
    bn_stats <<<1, 512, 0, stream>>>(colsum, colsq, gamma, beta, scale, shift);
    bn_apply <<<(N_NODES * DIM_H + 255) / 256, 256, 0, stream>>>(out, scale, shift);
}
